// FFTStackedHB_SegRes_48808008351931
// MI455X (gfx1250) — matrix operand order repaired and checked by reading, not compiled, not run on hardware
//
#include <hip/hip_runtime.h>
#include <hip/hip_bf16.h>

// MI455X (gfx1250) implementation.
// - fft/ifft channel filter  ->  per-segment 256x256 circulant GEMM on
//   v_wmma_f32_16x16x32_f16 matrix cores (C_s precomputed from Hr,Hi).
// - x / W1 pre-converted to f16, zero-padded K=784->800: the GEMM1 K-loop has
//   no guards/converts, and tiles are staged LDS-direct with
//   global_load_async_to_lds_b64 (ASYNCcnt) -- the CDNA5 async copy path.
// - h (33.5MB) stays L2-resident across the 20 sequential segments.
// Workspace requirement: ~89.3 MB.

typedef __attribute__((ext_vector_type(16))) _Float16 v16h;
typedef __attribute__((ext_vector_type(4)))  _Float16 v4h;
typedef __attribute__((ext_vector_type(8)))  float    v8f;
typedef __attribute__((ext_vector_type(4)))  float    v4f;

#define WMMA_F16(a, b, c) \
  __builtin_amdgcn_wmma_f32_16x16x32_f16(false, (a), false, (b), (short)0, (c), false, false)

#define B_ROWS 32768
#define D_IN   784
#define K_PAD  800     // 25 * 32
#define D_H    256
#define D_OUT  10
#define N_SEG  20

// Async copy: 8 bytes from global (per lane) directly into LDS, no VGPR round
// trip. ldsoff = byte offset inside the workgroup's LDS allocation (low 32
// bits of a generic pointer to __shared__), gaddr = 64-bit global VA.
__device__ __forceinline__ void async_g2l_b64(unsigned int ldsoff, const void* gp) {
  unsigned long long ga = (unsigned long long)(uintptr_t)gp;
  asm volatile("global_load_async_to_lds_b64 %0, %1, off" :: "v"(ldsoff), "v"(ga) : "memory");
}
__device__ __forceinline__ void wait_async0() {
  asm volatile("s_wait_asynccnt 0" ::: "memory");
}
__device__ __forceinline__ unsigned int lds_off(const void* p) {
  return (unsigned int)(uintptr_t)p;   // generic LDS pointer: low 32 bits = LDS offset
}

// ---------------------------------------------------------------------------
// x [B][784] f32 -> xh [B][800] f16 (zero-padded). One block per row.
// 784 = 196*4, so vec4 boundaries align with the pad boundary.
// ---------------------------------------------------------------------------
__global__ void k_xcast(const float* __restrict__ x, _Float16* __restrict__ xh) {
  const int row = blockIdx.x, t = threadIdx.x;
  if (t < 200) {
    v4h o = {};
    if (t < 196) {
      v4f v = *(const v4f*)(x + (size_t)row * D_IN + t * 4);
      o = __builtin_convertvector(v, v4h);
    }
    *(v4h*)(xh + (size_t)row * K_PAD + t * 4) = o;
  }
}

// W1 [784][256] f32 -> w1t [256][800] f16 (transposed, zero-padded).
__global__ void k_w1cast(const float* __restrict__ W1, _Float16* __restrict__ w1t) {
  const int c = blockIdx.x, t = threadIdx.x;
  for (int k = t; k < K_PAD; k += 256) {
    float v = (k < D_IN) ? W1[(size_t)k * D_H + c] : 0.f;
    w1t[(size_t)c * K_PAD + k] = (_Float16)v;
  }
}

// ---------------------------------------------------------------------------
// Circulant matrices: CT[s][n][k] = c_s[(n-k) mod 256], c_s = Re(ifft(H^50)).
// Stored [n][k] so WMMA B-fragments (col n = lane&15, K contiguous) are
// contiguous 32B per-lane loads.
// ---------------------------------------------------------------------------
__global__ void k_build_ct(const float* __restrict__ Hr, const float* __restrict__ Hi,
                           _Float16* __restrict__ CT) {
  __shared__ float Her[256], Hei[256], cc[256];
  const int s = blockIdx.x, t = threadIdx.x;
  float hr = Hr[s * 256 + t], hi = Hi[s * 256 + t];
  float r2 = hr * hr + hi * hi;
  float R  = powf(r2, 25.0f);                       // |H|^50
  float th = 50.0f * atan2f(hi, hr);
  Her[t] = R * cosf(th);
  Hei[t] = R * sinf(th);
  __syncthreads();
  const float w0 = 6.283185307179586f / 256.0f;
  float acc = 0.f;
  for (int mm = 0; mm < 256; ++mm) {
    float ang = w0 * (float)((mm * t) & 255);
    acc += Her[mm] * cosf(ang) - Hei[mm] * sinf(ang);
  }
  cc[t] = acc * (1.0f / 256.0f);
  __syncthreads();
  _Float16* row = CT + (size_t)(s * 256 + t) * 256;
  for (int k = 0; k < 256; ++k) row[k] = (_Float16)cc[(t - k) & 255];
}

// W2 [256][10] f32 -> W2p [16][256] f16 (n-major, zero-padded cols 10..15)
__global__ void k_w2pad(const float* __restrict__ W2, _Float16* __restrict__ W2p) {
  int n = blockIdx.x, k = threadIdx.x;
  float v = (n < D_OUT) ? W2[(size_t)k * D_OUT + n] : 0.f;
  W2p[n * 256 + k] = (_Float16)v;
}

// ---------------------------------------------------------------------------
// GEMM1: preact = xh @ w1t^T + b1  (32768 x 800 x 256), f16 WMMA, f32 accum.
// Block tile 64x64, 8 waves. Tiles staged with async-to-LDS (no guards, no
// converts in the K-loop).
// ---------------------------------------------------------------------------
__global__ void k_gemm1(const _Float16* __restrict__ xh, const _Float16* __restrict__ w1t,
                        const float* __restrict__ b1, float* __restrict__ out) {
  __shared__ _Float16 xs[64 * 40];   // A tile [row][k], stride 40 (80B rows)
  __shared__ _Float16 wl[64 * 40];   // B tile [col][k]
  const int tid = threadIdx.x;
  const int lane = tid & 31, wv = tid >> 5;
  const int r0 = blockIdx.x * 64, c0 = blockIdx.y * 64;
  const int rt = wv & 3, ctb = (wv >> 2) * 2;
  const int m = lane & 15, kh = lane >> 4;
  // async copy mapping: 512 x 8B chunks per tile, 2 per thread
  const int ea = tid, eb = tid + 256;
  const int ra = ea >> 3, ka = (ea & 7) * 4;       // xs: row, half-index
  const int rb = eb >> 3, kb = (eb & 7) * 4;
  const unsigned int lxa = lds_off(&xs[ra * 40 + ka]);
  const unsigned int lxb = lds_off(&xs[rb * 40 + kb]);
  const unsigned int lwa = lds_off(&wl[ra * 40 + ka]);
  const unsigned int lwb = lds_off(&wl[rb * 40 + kb]);
  const _Float16* gxa = xh + (size_t)(r0 + ra) * K_PAD + ka;
  const _Float16* gxb = xh + (size_t)(r0 + rb) * K_PAD + kb;
  const _Float16* gwa = w1t + (size_t)(c0 + ra) * K_PAD + ka;
  const _Float16* gwb = w1t + (size_t)(c0 + rb) * K_PAD + kb;
  v8f acc0 = {}, acc1 = {};
  for (int kk = 0; kk < 25; ++kk) {
    const int k0 = kk * 32;
    async_g2l_b64(lxa, gxa + k0);
    async_g2l_b64(lxb, gxb + k0);
    async_g2l_b64(lwa, gwa + k0);
    async_g2l_b64(lwb, gwb + k0);
    wait_async0();
    __syncthreads();
    v16h a, bb0, bb1;
#pragma unroll
    for (int j = 0; j < 8; ++j) {                  // A layout: K base by lane>=16 (contiguous, matches B)
      a[j]     = xs[(rt * 16 + m) * 40 + kh * 16 + j];
      a[j + 8] = xs[(rt * 16 + m) * 40 + kh * 16 + 8 + j];
    }
#pragma unroll
    for (int j = 0; j < 16; ++j) {                 // B layout: K base by lane>=16 (contiguous, matches A)
      bb0[j] = wl[((ctb + 0) * 16 + m) * 40 + kh * 16 + j];
      bb1[j] = wl[((ctb + 1) * 16 + m) * 40 + kh * 16 + j];
    }
    acc0 = WMMA_F16(a, bb0, acc0);
    acc1 = WMMA_F16(a, bb1, acc1);
    __syncthreads();
  }
#pragma unroll
  for (int r = 0; r < 8; ++r) {                    // C/D layout: M = r + 8*(lane>=16)
    int row = r0 + rt * 16 + kh * 8 + r;
    int c   = c0 + ctb * 16 + m;
    out[(size_t)row * D_H + c]      = acc0[r] + b1[c];
    out[(size_t)row * D_H + c + 16] = acc1[r] + b1[c + 16];
  }
}

// ---------------------------------------------------------------------------
// BN stats: deterministic coalesced 2-stage reduction (biased batch var).
// ---------------------------------------------------------------------------
__global__ void k_bnstat1(const float* __restrict__ a, float* __restrict__ ps,
                          float* __restrict__ pq) {
  const int b = blockIdx.x, t = threadIdx.x;       // block: 256 rows, thread: channel
  const float* p = a + (size_t)b * 256 * D_H + t;
  float s = 0.f, q = 0.f;
  for (int r = 0; r < 256; ++r) {
    float v = p[(size_t)r * D_H];                  // consecutive t -> coalesced
    s += v; q += v * v;
  }
  ps[b * D_H + t] = s;
  pq[b * D_H + t] = q;
}
__global__ void k_bnstat2(const float* __restrict__ ps, const float* __restrict__ pq,
                          float* __restrict__ mu, float* __restrict__ rsig) {
  const int t = threadIdx.x;
  float s = 0.f, q = 0.f;
  for (int b = 0; b < B_ROWS / 256; ++b) { s += ps[b * D_H + t]; q += pq[b * D_H + t]; }
  float mean = s * (1.0f / (float)B_ROWS);
  float var  = q * (1.0f / (float)B_ROWS) - mean * mean;
  mu[t]   = mean;
  rsig[t] = rsqrtf(var + 1e-5f);
}

__global__ void k_bnrelu(float* __restrict__ a, const float* __restrict__ mu,
                         const float* __restrict__ rsig, const float* __restrict__ g,
                         const float* __restrict__ be) {
  size_t i = (size_t)blockIdx.x * 256 + threadIdx.x;
  int ch = (int)(i & 255);
  float v = g[ch] * (a[i] - mu[ch]) * rsig[ch] + be[ch];
  a[i] = fmaxf(v, 0.f);
}

// ---------------------------------------------------------------------------
// One hyperbolic segment, fused: u=log0(h); v=u@C_s (WMMA); h=a*exp0(v)+b*h.
// Block: 32 rows x 256 cols, 8 waves; rows owned exclusively -> in-place safe.
// ---------------------------------------------------------------------------
#define US 272   // u LDS row stride (halfs, 544B rows)
#define VS 264   // v LDS row stride (floats, 1056B rows)
__global__ void k_segment(float* __restrict__ h, const _Float16* __restrict__ CTs,
                          const float* __restrict__ alphas, const float* __restrict__ betas,
                          int s) {
  __shared__ _Float16 U[32 * US];
  __shared__ float vbuf[32 * VS];
  __shared__ float red[256];
  __shared__ float scl[32];
  const float sc = 0.03162277660168379f;           // sqrt(C), C = 0.001
  const int tid = threadIdx.x, lane = tid & 31, wv = tid >> 5;
  const int r0 = blockIdx.x * 32;
  const int row = tid >> 3, part = tid & 7;        // 8 threads per row, 32 ch each

  // ---- log0: row norm + artanh scaling; stage u (f16) in LDS ----
  const v4f* hrd4 = (const v4f*)(h + (size_t)(r0 + row) * D_H + part * 32);
  v4f hv4[8];
  float sq = 0.f;
#pragma unroll
  for (int c = 0; c < 8; ++c) {
    hv4[c] = hrd4[c];
    sq += hv4[c][0] * hv4[c][0] + hv4[c][1] * hv4[c][1]
        + hv4[c][2] * hv4[c][2] + hv4[c][3] * hv4[c][3];
  }
  red[tid] = sq;
  __syncthreads();
  if (tid < 32) {
    float t2 = 0.f;
#pragma unroll
    for (int i = 0; i < 8; ++i) t2 += red[tid * 8 + i];
    float n  = sqrtf(t2);
    float sn = fminf(fmaxf(sc * n, 1e-12f), 1.0f - 1e-5f);
    float at = 0.5f * logf((1.f + sn) / (1.f - sn));   // artanh
    scl[tid] = at / sn;
  }
  __syncthreads();
  {
    float sl = scl[row];
#pragma unroll
    for (int c = 0; c < 8; ++c)
      *(v4h*)&U[row * US + part * 32 + c * 4] = __builtin_convertvector(hv4[c] * sl, v4h);
  }
  __syncthreads();

  // ---- v = u @ C_s: wave -> 1 row-tile x 4 col-tiles, K = 8x32 ----
  const int rt = wv >> 2;
  const int ct0 = (wv & 3) * 4;
  const int m = lane & 15, kh = lane >> 4;
  v16h af[8];
#pragma unroll
  for (int ks = 0; ks < 8; ++ks)
#pragma unroll
    for (int j = 0; j < 8; ++j) {
      af[ks][j]     = U[(rt * 16 + m) * US + ks * 32 + kh * 16 + j];
      af[ks][j + 8] = U[(rt * 16 + m) * US + ks * 32 + kh * 16 + 8 + j];
    }
  for (int c4 = 0; c4 < 4; ++c4) {
    const int ct = ct0 + c4;
    const _Float16* bcol = CTs + (size_t)(ct * 16 + m) * 256;
    __builtin_prefetch(bcol + 256, 0, 0);          // global_prefetch_b8 (next column)
    v8f acc = {};
#pragma unroll
    for (int ks = 0; ks < 8; ++ks) {
      v16h b;
      const _Float16* bp = bcol + ks * 32 + kh * 16;
#pragma unroll
      for (int j = 0; j < 16; ++j) b[j] = bp[j];
      acc = WMMA_F16(af[ks], b, acc);
    }
#pragma unroll
    for (int r = 0; r < 8; ++r)
      vbuf[(rt * 16 + kh * 8 + r) * VS + ct * 16 + m] = acc[r];
  }
  __syncthreads();

  // ---- exp0 + residual combine ----
  const v4f* vb4 = (const v4f*)&vbuf[row * VS + part * 32];
  float sqv = 0.f;
#pragma unroll
  for (int c = 0; c < 8; ++c) {
    v4f v = vb4[c];
    sqv += v[0] * v[0] + v[1] * v[1] + v[2] * v[2] + v[3] * v[3];
  }
  red[tid] = sqv;
  __syncthreads();
  if (tid < 32) {
    float t2 = 0.f;
#pragma unroll
    for (int i = 0; i < 8; ++i) t2 += red[tid * 8 + i];
    float n  = sqrtf(t2);
    float sn = fmaxf(sc * n, 1e-12f);
    scl[tid] = tanhf(sn) / sn;
  }
  __syncthreads();
  {
    float as = alphas[s] * scl[row];
    float bt = betas[s];
    v4f* hw4 = (v4f*)(h + (size_t)(r0 + row) * D_H + part * 32);
#pragma unroll
    for (int c = 0; c < 8; ++c) hw4[c] = vb4[c] * as + hv4[c] * bt;
  }
}

// ---------------------------------------------------------------------------
// Final: out = relu(BN2(h)) @ W2 + b2. BN fused into A-fragment conversion;
// N=10 padded to one 16-wide WMMA col-tile. Wave -> 16 rows, K = 8x32.
// ---------------------------------------------------------------------------
__global__ void k_gemm2bn(const float* __restrict__ h, const _Float16* __restrict__ W2p,
                          const float* __restrict__ mu, const float* __restrict__ rsig,
                          const float* __restrict__ g, const float* __restrict__ be,
                          const float* __restrict__ b2, float* __restrict__ out) {
  const int tid = threadIdx.x, lane = tid & 31, wv = tid >> 5;
  const int m = lane & 15, kh = lane >> 4;
  const int arow = blockIdx.x * 128 + wv * 16 + m;
  const float* hr = h + (size_t)arow * D_H;
  v8f acc = {};
  for (int ks = 0; ks < 8; ++ks) {
    v16h a, b;
#pragma unroll
    for (int j = 0; j < 8; ++j) {
      int c0 = ks * 32 + kh * 16 + j;
      int c1 = c0 + 8;
      float v0 = fmaxf(g[c0] * (hr[c0] - mu[c0]) * rsig[c0] + be[c0], 0.f);
      float v1 = fmaxf(g[c1] * (hr[c1] - mu[c1]) * rsig[c1] + be[c1], 0.f);
      a[j]     = (_Float16)v0;
      a[j + 8] = (_Float16)v1;
    }
    const _Float16* bp = W2p + m * 256 + ks * 32 + kh * 16;
#pragma unroll
    for (int j = 0; j < 16; ++j) b[j] = bp[j];
    acc = WMMA_F16(a, b, acc);
  }
  if (m < D_OUT) {
#pragma unroll
    for (int r = 0; r < 8; ++r) {
      int rr = blockIdx.x * 128 + wv * 16 + kh * 8 + r;
      out[(size_t)rr * D_OUT + m] = acc[r] + b2[m];
    }
  }
}

// ---------------------------------------------------------------------------
extern "C" void kernel_launch(void* const* d_in, const int* in_sizes, int n_in,
                              void* d_out, int out_size, void* d_ws, size_t ws_size,
                              hipStream_t stream) {
  const float* x   = (const float*)d_in[0];
  const float* W1  = (const float*)d_in[1];
  const float* b1  = (const float*)d_in[2];
  const float* g1  = (const float*)d_in[3];
  const float* be1 = (const float*)d_in[4];
  const float* Hr  = (const float*)d_in[5];
  const float* Hi  = (const float*)d_in[6];
  const float* al  = (const float*)d_in[7];
  const float* bt  = (const float*)d_in[8];
  const float* g2  = (const float*)d_in[9];
  const float* be2 = (const float*)d_in[10];
  const float* W2  = (const float*)d_in[11];
  const float* b2  = (const float*)d_in[12];
  float* out = (float*)d_out;

  // Workspace layout (every region written before read on every call):
  char* ws = (char*)d_ws;
  float*    hbuf = (float*)ws;                      //  0         : 33,554,432
  _Float16* xh   = (_Float16*)(ws + 33554432);      // +33,554,432: 52,428,800
  _Float16* w1t  = (_Float16*)(ws + 85983232);      // +85,983,232:    409,600
  _Float16* CT   = (_Float16*)(ws + 86392832);      // +86,392,832:  2,621,440
  _Float16* W2p  = (_Float16*)(ws + 89014272);      // +89,014,272:      8,192
  float*    ps   = (float*)(ws + 89022464);         //               131,072
  float*    pq   = (float*)(ws + 89153536);         //               131,072
  float*    mu1  = (float*)(ws + 89284608);
  float*    rs1  = mu1 + 256;
  float*    mu2  = rs1 + 256;
  float*    rs2  = mu2 + 256;

  hipLaunchKernelGGL(k_xcast,    dim3(B_ROWS), dim3(256), 0, stream, x, xh);
  hipLaunchKernelGGL(k_w1cast,   dim3(D_H),    dim3(256), 0, stream, W1, w1t);
  hipLaunchKernelGGL(k_build_ct, dim3(N_SEG),  dim3(256), 0, stream, Hr, Hi, CT);
  hipLaunchKernelGGL(k_w2pad,    dim3(16),     dim3(256), 0, stream, W2, W2p);
  hipLaunchKernelGGL(k_gemm1,    dim3(B_ROWS / 64, D_H / 64), dim3(256), 0, stream,
                     xh, w1t, b1, hbuf);
  hipLaunchKernelGGL(k_bnstat1,  dim3(B_ROWS / 256), dim3(256), 0, stream, hbuf, ps, pq);
  hipLaunchKernelGGL(k_bnstat2,  dim3(1),      dim3(256), 0, stream, ps, pq, mu1, rs1);
  hipLaunchKernelGGL(k_bnrelu,   dim3(B_ROWS * D_H / 256), dim3(256), 0, stream,
                     hbuf, mu1, rs1, g1, be1);
  for (int s = 0; s < N_SEG; ++s)
    hipLaunchKernelGGL(k_segment, dim3(B_ROWS / 32), dim3(256), 0, stream,
                       hbuf, CT + (size_t)s * 256 * 256, al, bt, s);
  hipLaunchKernelGGL(k_bnstat1,  dim3(B_ROWS / 256), dim3(256), 0, stream, hbuf, ps, pq);
  hipLaunchKernelGGL(k_bnstat2,  dim3(1),      dim3(256), 0, stream, ps, pq, mu2, rs2);
  hipLaunchKernelGGL(k_gemm2bn,  dim3(B_ROWS / 128), dim3(256), 0, stream,
                     hbuf, W2p, mu2, rs2, g2, be2, b2, out);
}
